// EdgeUpdateNet_64570538328261
// MI455X (gfx1250) — matrix operand order repaired and checked by reading, not compiled, not run on hardware
//
#include <hip/hip_runtime.h>
#include <hip/hip_bf16.h>
#include <math.h>

// ---------------------------------------------------------------------------
// EdgeUpdateNet (SchNet-like GNN) for gfx1250 / MI455X.
// Edge MLP GEMMs run on v_wmma_f32_16x16x32_bf16 (f32 accumulate).
// ---------------------------------------------------------------------------

#define C_DIM    64
#define NG_DIM   150
#define N_NODES  50000
#define N_EDGES  800000
#define N_GRAPHS 256

typedef __bf16 bf16_t;
typedef __attribute__((ext_vector_type(16))) __bf16 v16bf;
typedef __attribute__((ext_vector_type(8)))  float  v8f;

// POD 16-byte chunk (avoid HIP int4 ctor issues inside unions)
struct alignas(16) B16x8 { unsigned int w[4]; };
union Frag16 { v16bf v; bf16_t e[16]; B16x8 q[2]; };

__device__ __forceinline__ float sspf(float x) {
    // shifted softplus, numerically stable: max(x,0)+log1p(exp(-|x|)) - log(2)
    float ax = fabsf(x);
    return fmaxf(x, 0.0f) + __logf(1.0f + __expf(-ax)) - 0.69314718056f;
}

__device__ __forceinline__ void smear8(float d, int g0, bf16_t* out) {
    const float s = 0.31622776601683794f; // sqrt(0.1)
    #pragma unroll
    for (int j = 0; j < 8; ++j) {
        int g = g0 + j;
        float v = 0.0f;
        if (g < NG_DIM) { float t = d - 0.1f * (float)g; v = __expf(-s * t * t); }
        out[j] = (bf16_t)v;
    }
}

// ---------------------------------------------------------------------------
// Pre-swizzle W1 [Kin(pad to 32*kSteps) x 128] and W2 [128 x 64] (f32, global)
// into WMMA B-fragment order (bf16).  Fragment = 32 lanes x 16 bf16, stored
// contiguously per lane (32 B) so compute loads are b128s.
// B layout (16x16x32 bf16): lane L holds col n = L%16; elem i holds
//   K = 32*kt + (L<16 ? 0 : 8) + (i<8 ? i : 16 + (i-8)).
// ---------------------------------------------------------------------------
__global__ void prep_weights(const float* __restrict__ W1,
                             const float* __restrict__ W2,
                             bf16_t* __restrict__ w1f,
                             bf16_t* __restrict__ w2f,
                             int Kin, int kSteps) {
    int tid = blockIdx.x * blockDim.x + threadIdx.x;
    int totalW1 = kSteps * 8 * 32;          // frag-lane slots for W1
    if (tid < totalW1) {
        int lane = tid & 31, frag = tid >> 5;
        int kt = frag >> 3, nt = frag & 7;
        int hiL = lane >> 4, colL = lane & 15;
        bf16_t* dstp = w1f + (size_t)frag * 512 + lane * 16;
        #pragma unroll
        for (int i = 0; i < 16; ++i) {
            int k = kt * 32 + hiL * 8 + (i < 8 ? i : 16 + (i - 8));
            int n = nt * 16 + colL;
            float v = (k < Kin) ? W1[(size_t)k * 128 + n] : 0.0f;
            dstp[i] = (bf16_t)v;
        }
    } else {
        int t2 = tid - totalW1;
        if (t2 < 16 * 32) {
            int lane = t2 & 31, frag = t2 >> 5;
            int kt = frag >> 2, nt = frag & 3;
            int hiL = lane >> 4, colL = lane & 15;
            bf16_t* dstp = w2f + (size_t)frag * 512 + lane * 16;
            #pragma unroll
            for (int i = 0; i < 16; ++i) {
                int k = kt * 32 + hiL * 8 + (i < 8 ? i : 16 + (i - 8));
                int n = nt * 16 + colL;
                dstp[i] = (bf16_t)W2[(size_t)k * 64 + n];
            }
        }
    }
}

// ---------------------------------------------------------------------------
// Fused edge kernel: one wave per 16-edge tile.
//   pre1[16,128] = [h_src | h_dst | ea] @ W1 + b1      (bf16 WMMA, f32 acc)
//   ea'  [16, 64] = ssp(pre1) @ W2 + b2                (bf16 WMMA via LDS xpose)
//   ea_out = ea';  m[dst] += Hx[src] * ea'   (f32 atomics)
// A layout (16x16x32 bf16): lane L holds row m = L%16; elems {0..7,8..15}
//   cover K = 32*kt + (L<16?0:8) + {0..7, 16..23}  -> two contiguous 16B chunks.
// Templated on KSTEPS/USE_SMEAR so the per-kt source region (h_src / h_dst /
// ea|smear) is compile-time.  A 64-bit offset laundered through inline-asm is
// added to the weight-fragment addresses each tile: blocks LICM (which would
// hoist + spill the loads) while keeping them addrspace(1) global_load_b128s.
// All 8 B-frags of a kt are loaded into an array before the 8 WMMAs so the
// loads form one clause (one wait) and the WMMAs issue back-to-back.
// ---------------------------------------------------------------------------
template <int KSTEPS, int USE_SMEAR>
__global__ __launch_bounds__(256, 4) void edge_kernel(
    const int*    __restrict__ srcIdx,
    const int*    __restrict__ dstIdx,
    const float*  __restrict__ dist,
    const bf16_t* __restrict__ hB,     // [N,64] bf16
    const bf16_t* __restrict__ eaIn,   // [E,64] bf16 (layers >= 1)
    const bf16_t* __restrict__ w1f,    // swizzled W1 frags
    const bf16_t* __restrict__ w2f,    // swizzled W2 frags
    const float*  __restrict__ b1,     // [128]
    const float*  __restrict__ b2,     // [64]
    const float*  __restrict__ Hx,     // [N,64] f32 (h @ Wf + bf)
    bf16_t*       __restrict__ eaOut,  // [E,64] bf16
    float*        __restrict__ mOut) { // [N,64] f32, atomically accumulated
    __shared__ alignas(16) bf16_t scr[8][16 * 128]; // per-wave xpose scratch

    const int lane = threadIdx.x & 31;
    const int wave = threadIdx.x >> 5;
    const int wavesPerBlock = blockDim.x >> 5;
    const int col  = lane & 15;
    const int hi   = lane >> 4;
    const int nTiles = N_EDGES / 16;
    bf16_t* s = scr[wave];

    // tile-invariant bias values (C/D layout: lane holds col n = 16*nt + col)
    float bv1[8], bv2[4];
    #pragma unroll
    for (int nt = 0; nt < 8; ++nt) bv1[nt] = b1[nt * 16 + col];
    #pragma unroll
    for (int nt = 0; nt < 4; ++nt) bv2[nt] = b2[nt * 16 + col];

    size_t zoff = 0;   // always 0, but opaque to the optimizer (see below)

    for (int tile = blockIdx.x * wavesPerBlock + wave; tile < nTiles;
         tile += gridDim.x * wavesPerBlock) {
        // Opaque per-iteration offset: defeats LICM of the weight-fragment
        // loads without destroying the global address-space inference.
        asm volatile("" : "+s"(zoff));
        const bf16_t* w1p = w1f + zoff;
        const bf16_t* w2p = w2f + zoff;

        const int eBase = tile * 16;
        const int myE   = eBase + col;       // row this lane feeds for A-frags
        const int se    = srcIdx[myE];
        const int de    = dstIdx[myE];
        const float dmy = USE_SMEAR ? dist[myE] : 0.0f;
        const bf16_t* srow = hB + (size_t)se * 64;
        const bf16_t* drow = hB + (size_t)de * 64;

        // ----- GEMM1: pre1 = A @ W1 + b1 -----
        v8f acc[8];
        #pragma unroll
        for (int nt = 0; nt < 8; ++nt) {
            #pragma unroll
            for (int r = 0; r < 8; ++r) acc[nt][r] = bv1[nt];
        }
        #pragma unroll
        for (int kt = 0; kt < KSTEPS; ++kt) {
            Frag16 a;
            const int kb = kt * 32;          // compile-time region selector
            #pragma unroll
            for (int c2 = 0; c2 < 2; ++c2) {
                const int k0c = kb + c2 * 16;            // chunk base (c.t.)
                int k0 = k0c + hi * 8;                   // + lane offset
                bf16_t* dp = &a.e[c2 * 8];
                if (k0c < 64) {
                    *(B16x8*)dp = *(const B16x8*)(srow + k0);
                } else if (k0c < 128) {
                    *(B16x8*)dp = *(const B16x8*)(drow + (k0 - 64));
                } else if (USE_SMEAR) {
                    smear8(dmy, k0 - 128, dp);
                } else {
                    *(B16x8*)dp = *(const B16x8*)(eaIn + (size_t)myE * 64 + (k0 - 128));
                }
            }
            // load all 8 B-frags first (one clause, one wait), then 8 WMMAs
            const bf16_t* wrow = w1p + (size_t)(kt * 8) * 512 + lane * 16;
            Frag16 bf[8];
            #pragma unroll
            for (int nt = 0; nt < 8; ++nt) {
                bf[nt].q[0] = *(const B16x8*)(wrow + nt * 512);
                bf[nt].q[1] = *(const B16x8*)(wrow + nt * 512 + 8);
            }
            #pragma unroll
            for (int nt = 0; nt < 8; ++nt) {
                acc[nt] = __builtin_amdgcn_wmma_f32_16x16x32_bf16(
                    false, a.v, false, bf[nt].v, (short)0, acc[nt], false, false);
            }
        }

        // ----- ssp + transpose to row-major bf16 in LDS -----
        #pragma unroll
        for (int nt = 0; nt < 8; ++nt) {
            int n = nt * 16 + col;
            #pragma unroll
            for (int r = 0; r < 8; ++r) {
                int m = hi * 8 + r;          // C/D layout row for this VGPR
                s[m * 128 + n] = (bf16_t)sspf(acc[nt][r]);
            }
        }

        // ----- GEMM2: out = ssp(pre1) @ W2 + b2 -----
        v8f acc2[4];
        #pragma unroll
        for (int nt = 0; nt < 4; ++nt) {
            #pragma unroll
            for (int r = 0; r < 8; ++r) acc2[nt][r] = bv2[nt];
        }
        #pragma unroll
        for (int kt = 0; kt < 4; ++kt) {
            Frag16 a2;
            int kb = kt * 32 + hi * 8;
            a2.q[0] = *(const B16x8*)(s + col * 128 + kb);
            a2.q[1] = *(const B16x8*)(s + col * 128 + kb + 16);
            const bf16_t* wrow2 = w2p + (size_t)(kt * 4) * 512 + lane * 16;
            Frag16 bf2[4];
            #pragma unroll
            for (int nt = 0; nt < 4; ++nt) {
                bf2[nt].q[0] = *(const B16x8*)(wrow2 + nt * 512);
                bf2[nt].q[1] = *(const B16x8*)(wrow2 + nt * 512 + 8);
            }
            #pragma unroll
            for (int nt = 0; nt < 4; ++nt) {
                acc2[nt] = __builtin_amdgcn_wmma_f32_16x16x32_bf16(
                    false, a2.v, false, bf2[nt].v, (short)0, acc2[nt], false, false);
            }
        }

        // ----- epilogue: ea_out, msg = Hx[src]*ea', scatter-add to m[dst] -----
        #pragma unroll
        for (int r = 0; r < 8; ++r) {
            int m = hi * 8 + r;
            int sRow = __builtin_amdgcn_ds_bpermute(m * 4, se);
            int dRow = __builtin_amdgcn_ds_bpermute(m * 4, de);
            int e = eBase + m;
            #pragma unroll
            for (int nt = 0; nt < 4; ++nt) {
                int n = nt * 16 + col;
                float v = acc2[nt][r];
                eaOut[(size_t)e * 64 + n] = (bf16_t)v;
                float msg = Hx[(size_t)sRow * 64 + n] * v;
                atomicAdd(&mOut[(size_t)dRow * 64 + n], msg);
            }
        }
    }
}

// ---------------------------------------------------------------------------
// h init: h = embedding[z], h_bf16 = bf16(h)
// ---------------------------------------------------------------------------
__global__ __launch_bounds__(256) void k_embed(const int* __restrict__ z,
                                               const float* __restrict__ emb,
                                               float* __restrict__ h,
                                               bf16_t* __restrict__ hB) {
    size_t total = (size_t)N_NODES * 64;
    for (size_t i = blockIdx.x * (size_t)blockDim.x + threadIdx.x; i < total;
         i += (size_t)gridDim.x * blockDim.x) {
        int row = (int)(i >> 6), c = (int)(i & 63);
        float v = emb[(size_t)z[row] * 64 + c];
        h[i] = v;
        hB[i] = (bf16_t)v;
    }
}

// ---------------------------------------------------------------------------
// Small node-level GEMM: out[i,j] = act(sum_k in[i,k]*W[k,j] + b[j])
// optional residual accumulate into out (== h) and bf16 mirror update.
// K, NO <= 64; W cached in LDS. Used for fc1 / mlp1 stages (~0.4 GFLOP each).
// ---------------------------------------------------------------------------
__global__ __launch_bounds__(256) void node_gemm(const float* __restrict__ in,
                                                 const float* __restrict__ W,
                                                 const float* __restrict__ b,
                                                 float* __restrict__ out,
                                                 int Nrows, int K, int NO,
                                                 int act, int residual,
                                                 bf16_t* __restrict__ hBf) {
    __shared__ float sW[64 * 64];
    int nW = K * NO;
    for (int i = threadIdx.x; i < nW; i += blockDim.x) sW[i] = W[i];
    __syncthreads();
    size_t total = (size_t)Nrows * NO;
    for (size_t idx = blockIdx.x * (size_t)blockDim.x + threadIdx.x; idx < total;
         idx += (size_t)gridDim.x * blockDim.x) {
        int row = (int)(idx / NO), j = (int)(idx % NO);
        const float* ip = in + (size_t)row * K;
        float acc = b[j];
        for (int k = 0; k < K; ++k) acc = fmaf(ip[k], sW[k * NO + j], acc);
        if (act) acc = sspf(acc);
        if (residual) {
            acc += out[idx];
            if (hBf) hBf[idx] = (bf16_t)acc;
        }
        out[idx] = acc;
    }
}

// ---------------------------------------------------------------------------
// Readout: per node 64 -> ssp(32) -> 1, segment-sum over graphs via atomics.
// ---------------------------------------------------------------------------
__global__ __launch_bounds__(128) void k_readout(const float* __restrict__ h,
                                                 const int* __restrict__ batch,
                                                 const float* __restrict__ W1,
                                                 const float* __restrict__ b1,
                                                 const float* __restrict__ W2,
                                                 const float* __restrict__ b2,
                                                 float* __restrict__ out) {
    int row = blockIdx.x * blockDim.x + threadIdx.x;
    if (row >= N_NODES) return;
    const float* hr = h + (size_t)row * 64;
    float acc = b2[0];
    for (int j = 0; j < 32; ++j) {
        float t = b1[j];
        for (int k = 0; k < 64; ++k) t = fmaf(hr[k], W1[k * 32 + j], t);
        acc = fmaf(sspf(t), W2[j], acc);
    }
    atomicAdd(&out[batch[row]], acc);
}

// ---------------------------------------------------------------------------
// d_in flat order (setup_inputs insertion order):
//  0 z  1 edge_index  2 distances  3 batch  4 embedding
//  5..16  edge_params[0..2]{W1,b1,W2,b2}
// 17..34  inter_params[0..2]{Wf,bf,W1,b1,W2,b2}
// 35..38  out_params{W1,b1,W2,b2}
// ---------------------------------------------------------------------------
extern "C" void kernel_launch(void* const* d_in, const int* in_sizes, int n_in,
                              void* d_out, int out_size, void* d_ws, size_t ws_size,
                              hipStream_t stream) {
    (void)in_sizes; (void)n_in; (void)out_size; (void)ws_size;
    const int*   z     = (const int*)d_in[0];
    const int*   eidx  = (const int*)d_in[1];
    const float* dist  = (const float*)d_in[2];
    const int*   batch = (const int*)d_in[3];
    const float* emb   = (const float*)d_in[4];
    const int* src = eidx;
    const int* dst = eidx + N_EDGES;

    char* p = (char*)d_ws;
    auto alloc = [&](size_t bytes) -> char* {
        char* r = p; p += (bytes + 255) & ~(size_t)255; return r;
    };
    float*  h    = (float*)  alloc((size_t)N_NODES * 64 * 4);
    bf16_t* hB   = (bf16_t*) alloc((size_t)N_NODES * 64 * 2);
    float*  Hx   = (float*)  alloc((size_t)N_NODES * 64 * 4);
    float*  mbuf = (float*)  alloc((size_t)N_NODES * 64 * 4);
    bf16_t* ea   = (bf16_t*) alloc((size_t)N_EDGES * 64 * 2);
    bf16_t* w1f  = (bf16_t*) alloc((size_t)9 * 8 * 512 * 2);
    bf16_t* w2f  = (bf16_t*) alloc((size_t)16 * 512 * 2);

    hipMemsetAsync(d_out, 0, (size_t)N_GRAPHS * sizeof(float), stream);
    k_embed<<<2048, 256, 0, stream>>>(z, emb, h, hB);

    for (int l = 0; l < 3; ++l) {
        const float* eW1 = (const float*)d_in[5 + 4 * l];
        const float* eb1 = (const float*)d_in[6 + 4 * l];
        const float* eW2 = (const float*)d_in[7 + 4 * l];
        const float* eb2 = (const float*)d_in[8 + 4 * l];
        const float* iWf = (const float*)d_in[17 + 6 * l];
        const float* ibf = (const float*)d_in[18 + 6 * l];
        const float* iW1 = (const float*)d_in[19 + 6 * l];
        const float* ib1 = (const float*)d_in[20 + 6 * l];
        const float* iW2 = (const float*)d_in[21 + 6 * l];
        const float* ib2 = (const float*)d_in[22 + 6 * l];

        int Kin    = (l == 0) ? (NG_DIM + 128) : 192;   // 278 or 192
        int kSteps = (l == 0) ? 9 : 6;                  // K padded to 288 / 192
        int slots  = kSteps * 8 * 32 + 16 * 32;
        prep_weights<<<(slots + 255) / 256, 256, 0, stream>>>(eW1, eW2, w1f, w2f,
                                                              Kin, kSteps);
        // hx = h @ Wf + bf   (no activation)
        node_gemm<<<2048, 256, 0, stream>>>(h, iWf, ibf, Hx, N_NODES, 64, 64,
                                            0, 0, (bf16_t*)nullptr);
        hipMemsetAsync(mbuf, 0, (size_t)N_NODES * 64 * 4, stream);
        // ea' + scatter message accumulation
        if (l == 0) {
            edge_kernel<9, 1><<<2048, 256, 0, stream>>>(src, dst, dist, hB, ea,
                                                        w1f, w2f, eb1, eb2, Hx,
                                                        ea, mbuf);
        } else {
            edge_kernel<6, 0><<<2048, 256, 0, stream>>>(src, dst, dist, hB, ea,
                                                        w1f, w2f, eb1, eb2, Hx,
                                                        ea, mbuf);
        }
        // m -> mlp1 -> residual:  h += ssp(m@W1+b1)@W2+b2
        node_gemm<<<2048, 256, 0, stream>>>(mbuf, iW1, ib1, Hx, N_NODES, 64, 64,
                                            1, 0, (bf16_t*)nullptr);
        node_gemm<<<2048, 256, 0, stream>>>(Hx, iW2, ib2, h, N_NODES, 64, 64,
                                            0, 1, hB);
    }

    const float* oW1 = (const float*)d_in[35];
    const float* ob1 = (const float*)d_in[36];
    const float* oW2 = (const float*)d_in[37];
    const float* ob2 = (const float*)d_in[38];
    k_readout<<<(N_NODES + 127) / 128, 128, 0, stream>>>(h, batch, oW1, ob1,
                                                         oW2, ob2, (float*)d_out);
}
